// GNN_64476049047928
// MI455X (gfx1250) — compile-verified, matrix operand order checked by reading, not run on hardware
//
#include <hip/hip_runtime.h>

// Problem dims (reference): B=128, No=128, Nr=512, D=512
constexpr int kD   = 512;
constexpr int kNo  = 128;
constexpr long long kMObj = 128LL * 128;   // B*No = 16384 rows

typedef __attribute__((ext_vector_type(16))) __bf16        v16bf;
typedef __attribute__((ext_vector_type(8)))  float         v8f;
typedef __attribute__((ext_vector_type(8)))  unsigned int  v8u;
typedef __attribute__((ext_vector_type(4)))  unsigned int  u32x4;
typedef __attribute__((ext_vector_type(8)))  int           i32x8;
typedef __attribute__((ext_vector_type(4)))  int           i32x4;

// Pack two f32 into a bf16 pair (one dword): round-half-up then one v_perm_b32.
__device__ __forceinline__ unsigned packbf2(float a, float b) {
  union { float f; unsigned u; } xa, xb;
  xa.f = a; xb.f = b;
  return __builtin_amdgcn_perm(xb.u + 0x8000u, xa.u + 0x8000u, 0x07060302u);
}

// K-offset pattern of 16-bit WMMA A/B fragments (ISA 7.12.2)
__device__ __forceinline__ int kofs(int p, int h) {
  return ((p & 4) << 2) + ((p & 3) * 2) + h * 8;
}

// Build one 16x32 bf16 A-fragment from a row-major f32 row pointer.
// Per lane: two contiguous 8-float 32B-aligned chunks -> 4x b128 + 8x perm.
__device__ __forceinline__ v16bf load_a_frag(const float* __restrict__ ap, int h) {
  const float4* c0 = (const float4*)(ap + h * 8);
  const float4* c1 = (const float4*)(ap + 16 + h * 8);
  float4 q0 = c0[0], q1 = c0[1];
  float4 q2 = c1[0], q3 = c1[1];
  union { unsigned u[8]; v16bf v; } r;
  r.u[0] = packbf2(q0.x, q0.y);
  r.u[1] = packbf2(q0.z, q0.w);
  r.u[2] = packbf2(q1.x, q1.y);
  r.u[3] = packbf2(q1.z, q1.w);
  r.u[4] = packbf2(q2.x, q2.y);
  r.u[5] = packbf2(q2.z, q2.w);
  r.u[6] = packbf2(q3.x, q3.y);
  r.u[7] = packbf2(q3.z, q3.w);
  return r.v;
}

// Issue one TDM "superstep" copy: a 2-D tile of 2 x 8192 B rows, 32768 B apart
// in global (consecutive K-steps of the B-fragment stream), packed contiguously
// (16 KB) into LDS. D# per ISA ch.8: data_size=8B, tile_dim0=1024, tile_dim1=2,
// tensor_dim0_stride=4096 elements; tensor dims huge so nothing is OOB-clipped.
__device__ __forceinline__ void tdm_load_16k(unsigned lds_addr, unsigned long long ga) {
  u32x4 g0 = { 1u,                                   // count=1, user desc
               lds_addr,                             // lds_addr [63:32]
               (unsigned)ga,                         // global_addr lo
               (unsigned)((ga >> 32) & 0x1FFFFFFu) | 0x80000000u }; // hi | type=2
  i32x8 g1 = { 0x30000,                // data_size=3 (8B); no flags/pad/mask
               0,                      // abar=0, T0[15:0]=0           (T0=2^30)
               0x4000,                 // T0[31:16]=0x4000 | T1[15:0]=0 (T1=2^20)
               (int)(0x10u | (1024u << 16)), // T1[31:16]=0x10 | tile_dim0=1024
               2,                      // tile_dim1=2, tile_dim2=0
               4096,                   // tensor_dim0_stride = 4096 (x8B = 32768B)
               0, 0 };                 // stride0 hi / stride1
  i32x4 g2 = { 1, 1, 0, 0 };           // tensor_dim2=1, tensor_dim3=1
  i32x4 g3 = { 0, 1 << 16, 0, 0 };     // stride3=0, tensor_dim4=1, tile_dim4=0
#if __clang_major__ >= 23
  i32x8 gz = { 0, 0, 0, 0, 0, 0, 0, 0 };
  __builtin_amdgcn_tensor_load_to_lds(g0, g1, g2, g3, gz, 0);
#else
  __builtin_amdgcn_tensor_load_to_lds(g0, g1, g2, g3, 0);
#endif
}

// ---------------------------------------------------------------------------
// Weight pre-swizzle: f32 row-major (Ktot x 512) -> bf16 B-fragments.
// Fragment f = kblk*32 + nblk: 256 dwords, lane*8 + p, dword = bf16 pair (K,K+1).
// ---------------------------------------------------------------------------
__global__ void __launch_bounds__(256)
wconv_kernel(const float* __restrict__ W, unsigned* __restrict__ dst, int kblocks) {
  const int lane = threadIdx.x & 31;
  const int wave = threadIdx.x >> 5;
  const int f = blockIdx.x * 8 + wave;
  if (f >= kblocks * 32) return;
  const int kblk = f >> 5;
  const int nblk = f & 31;
  const int h = lane >> 4;
  const int n = nblk * 16 + (lane & 15);
  unsigned* out = dst + (size_t)f * 256 + (size_t)lane * 8;
#pragma unroll
  for (int p = 0; p < 8; ++p) {
    int k = kblk * 32 + kofs(p, h);
    float v0 = W[(size_t)k * kD + n];
    float v1 = W[(size_t)(k + 1) * kD + n];
    out[p] = packbf2(v0, v1);
  }
}

// ---------------------------------------------------------------------------
// Fused GEMM + epilogue; B-fragments streamed to LDS by the Tensor Data Mover.
//   MODE 0: attr = attr_vecs @ W_att + b                        -> attr_ws
//   MODE 1: new_attr = relu(obj@Wa0 + attr@Wa1 + b) + attr      -> out
//   MODE 2: new_rela = relu(obj[s]@Wr0 + rela@Wr1 + obj[o]@Wr2 + b) + rela, *mask
// WG: 8 waves, tile 128M x 128N. Wave: 32M x 64N = 2x4 wmma accums.
// B streaming: "superstep" = 2 K-steps = 16 KB per WG, 4-deep circular LDS
// buffer (64 KB), issued 3 supersteps ahead; s_wait_tensorcnt(2) => oldest done.
// One WG barrier per 16 WMMAs.
// ---------------------------------------------------------------------------
template <int MODE>
__global__ void __launch_bounds__(256)
gemm_kernel(const float* __restrict__ A0, const float* __restrict__ A1,
            const unsigned* __restrict__ Bfrag, const float* __restrict__ bias,
            const float* __restrict__ resid, const long long* __restrict__ edges,
            const float* __restrict__ mask, float* __restrict__ Cout) {
  const int lane  = threadIdx.x & 31;
  const int wave  = threadIdx.x >> 5;
  const int waveM = wave >> 1;          // 0..3
  const int waveN = wave & 1;           // 0..1
  const int h     = lane >> 4;
  const int ln    = lane & 15;
  const int nblk0 = blockIdx.x * 8 + waveN * 4;       // of 32 n-blocks
  const int m0    = blockIdx.y * 128 + waveM * 32;
  const int r0    = m0 + ln;
  const int r1    = m0 + 16 + ln;
  const int sstot = (MODE == 2) ? 24 : 16;            // total supersteps

  __shared__ __align__(128) unsigned char smemB[4 * 16384];
  // LDS aperture maps flat addr[31:0] -> LDS byte offset (ISA 10.2)
  const unsigned lds_base = (unsigned)(unsigned long long)(const void*)smemB;
  const unsigned char* Bslab0 =
      (const unsigned char*)Bfrag + (size_t)blockIdx.x * 8 * 1024;
  const bool issuer = (wave == 0);

  v8f acc[2][4];
#pragma unroll
  for (int s = 0; s < 2; ++s)
#pragma unroll
    for (int t = 0; t < 4; ++t)
      acc[s][t] = v8f{0.f, 0.f, 0.f, 0.f, 0.f, 0.f, 0.f, 0.f};

  // Prologue: fill the pipeline 3 supersteps deep (superstep stride = 64 KB).
  if (issuer) {
#pragma unroll
    for (int i = 0; i < 3; ++i)
      tdm_load_16k(lds_base + (unsigned)(i * 16384),
                   (unsigned long long)(Bslab0 + (size_t)i * 65536));
  }

  int gss = 0;  // global superstep across segments

  auto run_seg = [&](const float* ar0, const float* ar1, int nsteps) {
    for (int kb2 = 0; kb2 < nsteps / 2; ++kb2) {
      // Oldest in-flight slab (this superstep) complete; publish; refill.
      if (issuer) __builtin_amdgcn_s_wait_tensorcnt((short)2);
      __syncthreads();
      if (issuer && (gss + 3) < sstot)
        tdm_load_16k(lds_base + (unsigned)(((gss + 3) & 3) * 16384),
                     (unsigned long long)(Bslab0 + (size_t)(gss + 3) * 65536));

      const unsigned char* ssbase = smemB + ((gss & 3) * 16384) +
                                    (waveN * 4) * 1024 + lane * 32;
#pragma unroll
      for (int u = 0; u < 2; ++u) {
        // --- A fragments (2 x 16 rows): b128 loads + perm-pack to bf16 ---
        const int kb = kb2 * 2 + u;
        v16bf af0 = load_a_frag(ar0 + kb * 32, h);
        v16bf af1 = load_a_frag(ar1 + kb * 32, h);

        // --- B fragments from LDS: 32B/lane each (ds_load_b128 x2) ---
        v16bf bf4[4];
#pragma unroll
        for (int t = 0; t < 4; ++t) {
          union { v8u u; v16bf v; } bu;
          bu.u = *(const v8u*)(ssbase + u * 8192 + t * 1024);
          bf4[t] = bu.v;
        }

        // --- 8x v_wmma_f32_16x16x32_bf16 ---
#pragma unroll
        for (int t = 0; t < 4; ++t) {
          acc[0][t] = __builtin_amdgcn_wmma_f32_16x16x32_bf16(
              false, af0, false, bf4[t], (short)0, acc[0][t], false, false);
          acc[1][t] = __builtin_amdgcn_wmma_f32_16x16x32_bf16(
              false, af1, false, bf4[t], (short)0, acc[1][t], false, false);
        }
      }
      ++gss;
    }
  };

  if (MODE == 0) {
    // A = attr_vecs, lda = 2D = 1024, K = 1024 (32 steps)
    run_seg(A0 + (size_t)r0 * 1024, A0 + (size_t)r1 * 1024, 32);
  } else if (MODE == 1) {
    // A = [obj | attr], each segment K=512 (16 steps)
    run_seg(A0 + (size_t)r0 * kD, A0 + (size_t)r1 * kD, 16);
    run_seg(A1 + (size_t)r0 * kD, A1 + (size_t)r1 * kD, 16);
  } else {
    // A = [obj[s] | rela | obj[o]], gathered via int64 edges (+batch*No)
    long long e00 = edges[(size_t)r0 * 2 + 0];
    long long e01 = edges[(size_t)r0 * 2 + 1];
    long long e10 = edges[(size_t)r1 * 2 + 0];
    long long e11 = edges[(size_t)r1 * 2 + 1];
    int b0 = r0 >> 9;   // / Nr
    int b1 = r1 >> 9;
    const float* s0 = A0 + (size_t)(b0 * kNo + (int)e00) * kD;
    const float* s1 = A0 + (size_t)(b1 * kNo + (int)e10) * kD;
    const float* o0 = A0 + (size_t)(b0 * kNo + (int)e01) * kD;
    const float* o1 = A0 + (size_t)(b1 * kNo + (int)e11) * kD;
    run_seg(s0, s1, 16);
    run_seg(A1 + (size_t)r0 * kD, A1 + (size_t)r1 * kD, 16);
    run_seg(o0, o1, 16);
  }

  // --- epilogue: bias (+ relu + residual (+ mask)) ---
  // acc layout: lane L, dword j -> row m0 + s*16 + (L>>4)*8 + j, col = L&15
#pragma unroll
  for (int t = 0; t < 4; ++t) {
    const int col = (nblk0 + t) * 16 + ln;
    const float bv = bias[col];
#pragma unroll
    for (int s = 0; s < 2; ++s) {
#pragma unroll
      for (int j = 0; j < 8; ++j) {
        const int r = m0 + s * 16 + h * 8 + j;
        float v = acc[s][t][j] + bv;
        if (MODE == 0) {
          Cout[(size_t)r * kD + col] = v;
        } else {
          v = v > 0.f ? v : 0.f;
          v += resid[(size_t)r * kD + col];
          if (MODE == 2) v *= mask[r];
          Cout[(size_t)r * kD + col] = v;
        }
      }
    }
  }
}

// ---------------------------------------------------------------------------
extern "C" void kernel_launch(void* const* d_in, const int* in_sizes, int n_in,
                              void* d_out, int out_size, void* d_ws, size_t ws_size,
                              hipStream_t stream) {
  (void)in_sizes; (void)n_in; (void)out_size; (void)ws_size;
  const float*     obj    = (const float*)d_in[0];
  const float*     attrv  = (const float*)d_in[1];
  const float*     rela   = (const float*)d_in[2];
  const long long* edges  = (const long long*)d_in[3];
  const float*     mask   = (const float*)d_in[4];
  const float*     W_att  = (const float*)d_in[5];
  const float*     b_att  = (const float*)d_in[6];
  const float*     W_attr = (const float*)d_in[7];
  const float*     b_attr = (const float*)d_in[8];
  const float*     W_rela = (const float*)d_in[9];
  const float*     b_rela = (const float*)d_in[10];

  float* out_obj  = (float*)d_out;
  float* out_attr = out_obj + kMObj * kD;
  float* out_rela = out_attr + kMObj * kD;

  // workspace: attr f32 intermediate (33.5 MB) + bf16 weight fragments (3.7 MB)
  char*     ws      = (char*)d_ws;
  float*    attr_ws = (float*)ws;
  unsigned* wb_att  = (unsigned*)(ws + 33554432);
  unsigned* wb_attr = (unsigned*)(ws + 33554432 + 1048576);
  unsigned* wb_rela = (unsigned*)(ws + 33554432 + 2097152);

  // 1) weight f32 -> bf16 fragment swizzle
  wconv_kernel<<<128, 256, 0, stream>>>(W_att,  wb_att,  32);  // 1024x512
  wconv_kernel<<<128, 256, 0, stream>>>(W_attr, wb_attr, 32);  // 1024x512
  wconv_kernel<<<192, 256, 0, stream>>>(W_rela, wb_rela, 48);  // 1536x512

  // 2) new_obj passthrough
  hipMemcpyAsync(out_obj, obj, (size_t)kMObj * kD * sizeof(float),
                 hipMemcpyDeviceToDevice, stream);

  // 3) GEMM chain (stream-ordered: attr_ws produced before consumed)
  gemm_kernel<0><<<dim3(4, 128), 256, 0, stream>>>(
      attrv, nullptr, wb_att, b_att, nullptr, nullptr, nullptr, attr_ws);
  gemm_kernel<1><<<dim3(4, 128), 256, 0, stream>>>(
      obj, attr_ws, wb_attr, b_attr, attr_ws, nullptr, nullptr, out_attr);
  gemm_kernel<2><<<dim3(4, 512), 256, 0, stream>>>(
      obj, rela, wb_rela, b_rela, rela, edges, mask, out_rela);
}
